// DecoderV2_57157424775293
// MI455X (gfx1250) — compile-verified, matrix operand order checked by reading, not run on hardware
//
#include <hip/hip_runtime.h>
#include <stdint.h>

#define B_ 32
#define T_ 64
#define S_ 400
#define EMB_ 512
#define DEC_ 512
#define VOCAB_ 32000
#define NEG_ 1e12f

typedef __bf16 bf16_t;
typedef bf16_t v16bf  __attribute__((ext_vector_type(16)));
typedef bf16_t bf16x8 __attribute__((ext_vector_type(8)));
typedef float  v8f    __attribute__((ext_vector_type(8)));

// ---------- monotonic float<->uint mapping for atomicMax on floats ----------
__device__ __forceinline__ unsigned fmap(float f) {
    unsigned u = __float_as_uint(f);
    return (u & 0x80000000u) ? ~u : (u | 0x80000000u);
}
__device__ __forceinline__ float funmap(unsigned m) {
    unsigned u = (m & 0x80000000u) ? (m ^ 0x80000000u) : ~m;
    return __uint_as_float(u);
}

// ---------- fp32 -> bf16 conversion (one-time; keeps layout) ----------
__global__ void cvt_bf16(const float* __restrict__ in, bf16_t* __restrict__ out, int n) {
    for (int i = blockIdx.x * blockDim.x + threadIdx.x; i < n; i += gridDim.x * blockDim.x)
        out[i] = (bf16_t)in[i];
}

// ---------- load one 16x32 fragment half-row: two contiguous 16B chunks ----------
__device__ __forceinline__ v16bf load_frag(const bf16_t* __restrict__ p) {
    bf16x8 lo = *(const bf16x8*)(p);
    bf16x8 hi = *(const bf16x8*)(p + 16);
    return __builtin_shufflevector(lo, hi, 0, 1, 2, 3, 4, 5, 6, 7,
                                           8, 9, 10, 11, 12, 13, 14, 15);
}

// ---------------------------------------------------------------------------
// bf16 WMMA GEMM: out[M,N] = A[M,K](bf16) * B^T (B stored bf16 [N][K]) + bias
// One wave per N-tile covering BOTH M-tiles of an M-pair (32 rows).
// MODE 0: f32 store (+ optional bf16 mirror via out_bf).
// MODE 1: maxout over adjacent col pairs -> bf16 out_bf [M,N/2].
// MODE 2: logit epilogue: + pointer scat (read, unmap, reset), ==0 -> -NEG,
//         store to d_out logits[b][t][v].
// ---------------------------------------------------------------------------
template <int MODE>
__global__ __launch_bounds__(256) void gemm_wmma(
    const bf16_t* __restrict__ A, int lda,
    const bf16_t* __restrict__ Bm, int ldb,
    const float* __restrict__ bias,
    float* __restrict__ out, bf16_t* __restrict__ out_bf,
    int Mpairs, int Ntiles, int K,
    unsigned* scat, int t, int VEXT)
{
    const int lane = threadIdx.x & 31;
    const int gw   = blockIdx.x * 8 + (threadIdx.x >> 5);
    if (gw >= Mpairs * Ntiles) return;                 // wave-uniform: EXEC stays all-ones
    const int nt   = gw % Ntiles;
    const int mp   = gw / Ntiles;
    const int half = lane >> 4;
    const int l16  = lane & 15;
    const int coln = nt * 16 + l16;

    const bf16_t* brow = Bm + (size_t)coln * ldb;
    const bf16_t* ar0  = A + (size_t)(mp * 32 + l16) * lda;
    const bf16_t* ar1  = ar0 + (size_t)16 * lda;

    v8f c0 = {}; v8f c1 = {};
    for (int k0 = 0; k0 < K; k0 += 32) {
        const int off = k0 + half * 8;
        v16bf bf = load_frag(brow + off);
        v16bf a0 = load_frag(ar0 + off);
        v16bf a1 = load_frag(ar1 + off);
        c0 = __builtin_amdgcn_wmma_f32_16x16x32_bf16(false, a0, false, bf, (short)0, c0, false, false);
        c1 = __builtin_amdgcn_wmma_f32_16x16x32_bf16(false, a1, false, bf, (short)0, c1, false, false);
    }

    const float bv = bias[coln];

    if (MODE == 0) {
        const int ldo = Ntiles * 16;
#pragma unroll
        for (int g = 0; g < 8; ++g) {
            const int m  = g + 8 * half;
            const float v0 = c0[g] + bv;
            const float v1 = c1[g] + bv;
            out[(size_t)(mp * 32 + m) * ldo + coln]      = v0;
            out[(size_t)(mp * 32 + 16 + m) * ldo + coln] = v1;
            if (out_bf) {
                out_bf[(size_t)(mp * 32 + m) * ldo + coln]      = (bf16_t)v0;
                out_bf[(size_t)(mp * 32 + 16 + m) * ldo + coln] = (bf16_t)v1;
            }
        }
    } else if (MODE == 1) {
        const int ldo = Ntiles * 8;   // N/2 after maxout
#pragma unroll
        for (int g = 0; g < 8; ++g) {
            float v0 = c0[g] + bv, v1 = c1[g] + bv;
            float p0 = __shfl_xor(v0, 1, 32);
            float p1 = __shfl_xor(v1, 1, 32);
            if ((l16 & 1) == 0) {
                const int m  = g + 8 * half;
                const int cc = coln >> 1;
                out_bf[(size_t)(mp * 32 + m) * ldo + cc]      = (bf16_t)fmaxf(v0, p0);
                out_bf[(size_t)(mp * 32 + 16 + m) * ldo + cc] = (bf16_t)fmaxf(v1, p1);
            }
        }
    } else { // MODE 2: logit + pointer merge, self-resetting scat
        const unsigned NEGM = fmap(-NEG_);
#pragma unroll
        for (int g = 0; g < 8; ++g) {
            const int b0i = mp * 32 + g + 8 * half;   // batch row (Mpairs==1 -> 0..15)
            const int b1i = b0i + 16;
            {
                const size_t si = (size_t)b0i * VEXT + coln;
                unsigned raw = scat[si]; scat[si] = NEGM;
                float s = funmap(raw); if (s == -NEG_) s = 0.f;
                float val = c0[g] + bv + s; if (val == 0.f) val = -NEG_;
                out[((size_t)b0i * T_ + t) * VEXT + coln] = val;
            }
            {
                const size_t si = (size_t)b1i * VEXT + coln;
                unsigned raw = scat[si]; scat[si] = NEGM;
                float s = funmap(raw); if (s == -NEG_) s = 0.f;
                float val = c1[g] + bv + s; if (val == 0.f) val = -NEG_;
                out[((size_t)b1i * T_ + t) * VEXT + coln] = val;
            }
        }
    }
}

// ---------- init: coverage=0, ctx slot of x_cat_bf=0, scat=-NEG (mapped) ----------
__global__ void init_kernel(float* cov, bf16_t* x_cat_bf, unsigned* scat, int VEXT) {
    const unsigned NEGM = fmap(-NEG_);
    const int stride = gridDim.x * blockDim.x;
    const int tid = blockIdx.x * blockDim.x + threadIdx.x;
    for (size_t i = tid; i < (size_t)B_ * VEXT; i += stride) scat[i] = NEGM;
    for (int i = tid; i < B_ * S_; i += stride) cov[i] = 0.f;
    for (int i = tid; i < B_ * 512; i += stride) {
        int b = i >> 9, j = i & 511;
        x_cat_bf[b * 1024 + 512 + j] = (bf16_t)0.f;   // pre_context c0 = 0
    }
}

// ---------- per-step: copy emb_t into x_cat_bf[:, 0:512] ----------
__global__ void prep_kernel(const float* __restrict__ trg, bf16_t* __restrict__ x_cat_bf, int t) {
    int i = blockIdx.x * blockDim.x + threadIdx.x;
    if (i < B_ * 512) {
        int b = i >> 9, j = i & 511;
        x_cat_bf[b * 1024 + j] = (bf16_t)trg[((size_t)b * T_ + t) * EMB_ + j];
    }
}

// ---------- GRU gate combine: h <- GRUCell(gi, gh, h) in place, + bf16 mirror ----------
__global__ void gru_combine(const float* __restrict__ gi, const float* __restrict__ gh,
                            float* __restrict__ h, bf16_t* __restrict__ h_bf) {
    int i = blockIdx.x * blockDim.x + threadIdx.x;
    if (i >= B_ * 512) return;
    int b = i >> 9, j = i & 511;
    const float* gib = gi + b * 1536;
    const float* ghb = gh + b * 1536;
    float r = 1.f / (1.f + __expf(-(gib[j] + ghb[j])));
    float z = 1.f / (1.f + __expf(-(gib[512 + j] + ghb[512 + j])));
    float n = tanhf(gib[1024 + j] + r * ghb[1024 + j]);
    float hv = (1.f - z) * n + z * h[i];
    h[i] = hv;
    h_bf[i] = (bf16_t)hv;
}

// ---------- attention: query GEMV, energy, masked softmax, context, coverage ----------
__global__ __launch_bounds__(256) void attn_kernel(
    int t, const float* __restrict__ h, const float* __restrict__ pre,
    float* __restrict__ cov, const float* __restrict__ memories,
    const float* __restrict__ W_q, const float* __restrict__ W_cov,
    const float* __restrict__ v_att, const unsigned char* __restrict__ mask,
    const float* __restrict__ trg, float* __restrict__ energy_ws,
    bf16_t* __restrict__ x_ro_bf, bf16_t* __restrict__ x_cat_bf,
    float* __restrict__ attn_out, float* __restrict__ cov_out)
{
    const int b = blockIdx.x, tid = threadIdx.x;
    __shared__ float sh_h[512], sh_t[512], sh_e[S_], sh_sc[S_], red[256];

    sh_h[tid]       = h[b * 512 + tid];
    sh_h[tid + 256] = h[b * 512 + tid + 256];
    __syncthreads();

    for (int j = tid; j < 512; j += 256) {           // target = h @ W_q^T
        const float* wq = W_q + (size_t)j * 512;
        float acc = 0.f;
        for (int k = 0; k < 512; ++k) acc += sh_h[k] * wq[k];
        sh_t[j] = acc;
    }
    __syncthreads();

    for (int s = tid; s < S_; s += 256) {            // energy
        float cv = cov[b * S_ + s];
        const float* pr = pre + ((size_t)b * S_ + s) * 512;
        float e = 0.f;
        for (int k = 0; k < 512; ++k)
            e += v_att[k] * tanhf(pr[k] + sh_t[k] + cv * W_cov[k]);
        if (mask[b * S_ + s]) e = -NEG_;
        sh_e[s] = e;
        energy_ws[b * S_ + s] = e;
    }
    __syncthreads();

    float m = -3.4e38f;                              // softmax max
    for (int s = tid; s < S_; s += 256) m = fmaxf(m, sh_e[s]);
    red[tid] = m; __syncthreads();
    for (int o = 128; o > 0; o >>= 1) { if (tid < o) red[tid] = fmaxf(red[tid], red[tid + o]); __syncthreads(); }
    const float mx = red[0]; __syncthreads();

    float sum = 0.f;                                 // softmax sum
    for (int s = tid; s < S_; s += 256) { float ev = __expf(sh_e[s] - mx); sh_sc[s] = ev; sum += ev; }
    red[tid] = sum; __syncthreads();
    for (int o = 128; o > 0; o >>= 1) { if (tid < o) red[tid] += red[tid + o]; __syncthreads(); }
    const float inv = 1.f / red[0]; __syncthreads();

    for (int s = tid; s < S_; s += 256) {            // score, history, coverage update
        float sc = sh_sc[s] * inv; sh_sc[s] = sc;
        float cvold = cov[b * S_ + s];
        attn_out[((size_t)t * B_ + b) * S_ + s] = sc;
        cov_out[((size_t)t * B_ + b) * S_ + s] = cvold;   // coverage BEFORE update
        cov[b * S_ + s] = cvold + sc;
    }
    __syncthreads();

    for (int e2 = tid; e2 < 512; e2 += 256) {        // context = score @ memories
        float acc = 0.f;
        for (int s = 0; s < S_; ++s) acc += sh_sc[s] * memories[((size_t)b * S_ + s) * 512 + e2];
        x_ro_bf[b * 1536 + 1024 + e2] = (bf16_t)acc; // readout input (ctx)
        x_cat_bf[b * 1024 + 512 + e2] = (bf16_t)acc; // GRU input for next step
    }
    for (int j = tid; j < 512; j += 256) {           // readout input (emb, h)
        x_ro_bf[b * 1536 + j]       = (bf16_t)trg[((size_t)b * T_ + t) * EMB_ + j];
        x_ro_bf[b * 1536 + 512 + j] = (bf16_t)sh_h[j];
    }
}

// ---------- pointer scatter-max into extended vocab (monotonic-uint atomicMax) ----------
__global__ void scatter_kernel(const int* __restrict__ ext_ids, const float* __restrict__ energy,
                               unsigned* __restrict__ scat, int VEXT) {
    const int b = blockIdx.x;
    for (int s = threadIdx.x; s < S_; s += blockDim.x) {
        int id = ext_ids[b * S_ + s];
        atomicMax(&scat[(size_t)b * VEXT + id], fmap(energy[b * S_ + s]));
    }
}

// ---------- OOV region of the logits (v >= VOCAB): ext_logit is 0 ----------
__global__ void ext_kernel(unsigned* __restrict__ scat, float* __restrict__ logits, int t, int VEXT) {
    const int W = VEXT - VOCAB_;
    if (W <= 0) return;
    const unsigned NEGM = fmap(-NEG_);
    const int n = B_ * W;
    for (int i = blockIdx.x * blockDim.x + threadIdx.x; i < n; i += gridDim.x * blockDim.x) {
        int b = i / W, v = VOCAB_ + i % W;
        size_t si = (size_t)b * VEXT + v;
        unsigned raw = scat[si]; scat[si] = NEGM;
        float s = funmap(raw); if (s == -NEG_) s = 0.f;
        float val = s; if (val == 0.f) val = -NEG_;
        logits[((size_t)b * T_ + t) * VEXT + v] = val;
    }
}

// ===========================================================================
extern "C" void kernel_launch(void* const* d_in, const int* in_sizes, int n_in,
                              void* d_out, int out_size, void* d_ws, size_t ws_size,
                              hipStream_t stream)
{
    const float* trg        = (const float*)d_in[0];
    const float* enc_states = (const float*)d_in[1];
    const float* memories   = (const float*)d_in[2];
    const float* W_trans    = (const float*)d_in[3];
    const float* b_trans    = (const float*)d_in[4];
    const float* W_ih       = (const float*)d_in[5];
    const float* W_hh       = (const float*)d_in[6];
    const float* b_ih       = (const float*)d_in[7];
    const float* b_hh       = (const float*)d_in[8];
    const float* W_pre      = (const float*)d_in[9];
    const float* b_pre      = (const float*)d_in[10];
    const float* W_q        = (const float*)d_in[11];
    const float* W_cov      = (const float*)d_in[12];
    const float* v_att      = (const float*)d_in[13];
    const float* W_ro       = (const float*)d_in[14];
    const float* b_ro       = (const float*)d_in[15];
    const float* W_logit    = (const float*)d_in[16];
    const float* b_logit    = (const float*)d_in[17];
    const int*   ext_ids    = (const int*)d_in[18];
    const unsigned char* mask = (const unsigned char*)d_in[19];

    int VEXT = (out_size - 2 * T_ * B_ * S_) / (B_ * T_);
    if (VEXT < VOCAB_) VEXT = VOCAB_;
    float* logits   = (float*)d_out;
    float* attn_out = logits + (size_t)B_ * T_ * VEXT;
    float* cov_out  = attn_out + (size_t)T_ * B_ * S_;

    // ---- workspace layout ----
    char* ws = (char*)d_ws; size_t off = 0;
    auto take = [&](size_t bytes) -> char* {
        char* p = ws + off; off = (off + bytes + 255) & ~(size_t)255; return p;
    };
    bf16_t* wih_bf    = (bf16_t*)take(sizeof(bf16_t) * 1536 * 1024);
    bf16_t* whh_bf    = (bf16_t*)take(sizeof(bf16_t) * 1536 * 512);
    bf16_t* wro_bf    = (bf16_t*)take(sizeof(bf16_t) * 512 * 1536);
    bf16_t* wlog_bf   = (bf16_t*)take(sizeof(bf16_t) * VOCAB_ * 256);
    bf16_t* wtrans_bf = (bf16_t*)take(sizeof(bf16_t) * 512 * 512);
    bf16_t* wpre_bf   = (bf16_t*)take(sizeof(bf16_t) * 512 * 512);
    bf16_t* enc_bf    = (bf16_t*)take(sizeof(bf16_t) * B_ * 512);
    bf16_t* mem_bf    = (bf16_t*)take(sizeof(bf16_t) * (size_t)B_ * S_ * 512);
    bf16_t* x_cat_bf  = (bf16_t*)take(sizeof(bf16_t) * B_ * 1024);
    bf16_t* x_ro_bf   = (bf16_t*)take(sizeof(bf16_t) * B_ * 1536);
    bf16_t* h_bf      = (bf16_t*)take(sizeof(bf16_t) * B_ * 512);
    bf16_t* mo_bf     = (bf16_t*)take(sizeof(bf16_t) * B_ * 256);
    float*  pre    = (float*)take(sizeof(float) * (size_t)B_ * S_ * 512);
    float*  h      = (float*)take(sizeof(float) * B_ * 512);
    float*  gi     = (float*)take(sizeof(float) * B_ * 1536);
    float*  gh     = (float*)take(sizeof(float) * B_ * 1536);
    float*  cov    = (float*)take(sizeof(float) * B_ * S_);
    float*  energy = (float*)take(sizeof(float) * B_ * S_);
    unsigned* scat = (unsigned*)take(sizeof(unsigned) * (size_t)B_ * VEXT);

    auto nblk = [](int tiles) { return (tiles + 7) / 8; };

    // ---- one-time setup ----
    cvt_bf16<<<512, 256, 0, stream>>>(W_ih,    wih_bf,    1536 * 1024);
    cvt_bf16<<<512, 256, 0, stream>>>(W_hh,    whh_bf,    1536 * 512);
    cvt_bf16<<<512, 256, 0, stream>>>(W_ro,    wro_bf,    512 * 1536);
    cvt_bf16<<<512, 256, 0, stream>>>(W_logit, wlog_bf,   VOCAB_ * 256);
    cvt_bf16<<<512, 256, 0, stream>>>(W_trans, wtrans_bf, 512 * 512);
    cvt_bf16<<<512, 256, 0, stream>>>(W_pre,   wpre_bf,   512 * 512);
    cvt_bf16<<<64,  256, 0, stream>>>(enc_states, enc_bf, B_ * 512);
    cvt_bf16<<<512, 256, 0, stream>>>(memories, mem_bf,   B_ * S_ * 512);

    // h0 = enc_states @ W_trans^T + b_trans   [32,512] (+ bf16 mirror)
    gemm_wmma<0><<<nblk(1 * 32), 256, 0, stream>>>(enc_bf, 512, wtrans_bf, 512,
                                                   b_trans, h, h_bf, 1, 32, 512, nullptr, 0, 0);
    // precompute = memories @ W_pre^T + b_pre   [32*400, 512]
    gemm_wmma<0><<<nblk(400 * 32), 256, 0, stream>>>(mem_bf, 512, wpre_bf, 512,
                                                     b_pre, pre, nullptr, 400, 32, 512, nullptr, 0, 0);
    init_kernel<<<512, 256, 0, stream>>>(cov, x_cat_bf, scat, VEXT);

    // ---- decode loop ----
    for (int t = 0; t < T_; ++t) {
        prep_kernel<<<64, 256, 0, stream>>>(trg, x_cat_bf, t);
        // gi = [emb, ctx] @ W_ih^T + b_ih   (M=32, N=1536, K=1024)
        gemm_wmma<0><<<nblk(96), 256, 0, stream>>>(x_cat_bf, 1024, wih_bf, 1024,
                                                   b_ih, gi, nullptr, 1, 96, 1024, nullptr, 0, 0);
        // gh = h @ W_hh^T + b_hh            (M=32, N=1536, K=512)
        gemm_wmma<0><<<nblk(96), 256, 0, stream>>>(h_bf, 512, whh_bf, 512,
                                                   b_hh, gh, nullptr, 1, 96, 512, nullptr, 0, 0);
        gru_combine<<<64, 256, 0, stream>>>(gi, gh, h, h_bf);
        attn_kernel<<<32, 256, 0, stream>>>(t, h, pre, cov, memories, W_q, W_cov, v_att,
                                            mask, trg, energy, x_ro_bf, x_cat_bf, attn_out, cov_out);
        scatter_kernel<<<32, 256, 0, stream>>>(ext_ids, energy, scat, VEXT);
        if (VEXT > VOCAB_) ext_kernel<<<8, 256, 0, stream>>>(scat, logits, t, VEXT);
        // mo = maxout(x_ro @ W_ro^T + b_ro)  (M=32, N=512 -> 256 bf16, K=1536)
        gemm_wmma<1><<<nblk(32), 256, 0, stream>>>(x_ro_bf, 1536, wro_bf, 1536,
                                                   b_ro, nullptr, mo_bf, 1, 32, 1536, nullptr, 0, 0);
        // logits[:, t, :VOCAB] = mo @ W_logit^T + b_logit + pointer merge
        gemm_wmma<2><<<nblk(VOCAB_ / 16), 256, 0, stream>>>(mo_bf, 256, wlog_bf, 256,
                                                            b_logit, logits, nullptr,
                                                            1, VOCAB_ / 16, 256,
                                                            scat, t, VEXT);
    }
    (void)in_sizes; (void)n_in; (void)ws_size;
}
